// AffineFlow_30399778521402
// MI455X (gfx1250) — matrix operand order repaired and checked by reading, not compiled, not run on hardware
//
#include <hip/hip_runtime.h>
#include <hip/hip_bf16.h>
#include <math.h>

// ---------------- problem constants (from reference) ----------------
#define BROWS 8192
#define CDIM  512
#define DD1   256
#define DD2   256
#define COUPW 1024
#define NBLK  2
#define NFLOW 8

#define MT    64          // rows per workgroup (4 x 16-row WMMA blocks)
#define NTHR  256         // 8 waves of 32; each wave owns all 64 rows x 64 cols

typedef __attribute__((ext_vector_type(8)))  __bf16 v8bf;
typedef __attribute__((ext_vector_type(16))) __bf16 v16bf;
typedef __attribute__((ext_vector_type(8)))  float  v8f;

// ---------------- WMMA helpers ----------------
// 4 M-blocks x one 16-col N-tile, accumulated over K.
// One B-fragment (global, L2-resident bf16 weights, W[n][k] row-major over k)
// is reused by 4 WMMAs against 4 A-fragments from LDS -> 4x less L2 traffic.
__device__ __forceinline__ void gemm_tile4(const __bf16* __restrict__ A, int lda,
                                           const __bf16* __restrict__ Bw, int ldb,
                                           int K, int lane, v8f acc[4]) {
  const int l15   = lane & 15;
  const int ahalf = (lane < 16) ? 0 : 8;    // A lanes: K 0..7/16..23 vs 8..15/24..31
  const int bhalf = (lane < 16) ? 0 : 8;    // B lanes: K 0..7/16..23 vs 8..15/24..31
  const __bf16* ap = A  + l15 * lda + ahalf;
  const __bf16* bp = Bw + l15 * ldb + bhalf;
  for (int k0 = 0; k0 < K; k0 += 32) {
    v8bf blo = *(const v8bf*)(bp + k0);
    v8bf bhi = *(const v8bf*)(bp + k0 + 16);
    __builtin_prefetch(bp + k0 + 1024, 0, 1);
    v16bf Bf = __builtin_shufflevector(blo, bhi, 0,1,2,3,4,5,6,7,8,9,10,11,12,13,14,15);
#pragma unroll
    for (int m = 0; m < 4; ++m) {
      const __bf16* am = ap + m * 16 * lda;
      v8bf alo = *(const v8bf*)(am + k0);
      v8bf ahi = *(const v8bf*)(am + k0 + 16);
      v16bf Af = __builtin_shufflevector(alo, ahi, 0,1,2,3,4,5,6,7,8,9,10,11,12,13,14,15);
      acc[m] = __builtin_amdgcn_wmma_f32_16x16x32_bf16(
          false, Af, false, Bf, (short)0, acc[m], false, false);
    }
  }
}

// D-matrix layout: VGPR i, lanes 0-15 -> (M=i, N=lane); lanes 16-31 -> (M=i+8).
__device__ __forceinline__ void store_tile(__bf16* __restrict__ dst, int ldd,
                                           v8f acc, int lane,
                                           float bias, float postmul, bool relu) {
  const int col   = lane & 15;
  const int rbase = (lane < 16) ? 0 : 8;
#pragma unroll
  for (int i = 0; i < 8; ++i) {
    float v = (acc[i] + bias) * postmul;
    if (relu) v = fmaxf(v, 0.f);
    dst[(rbase + i) * ldd + col] = (__bf16)v;
  }
}

__device__ __forceinline__ void store_tile4(__bf16* __restrict__ dst, int ldd,
                                            const v8f acc[4], int lane,
                                            float bias, float postmul, bool relu) {
#pragma unroll
  for (int m = 0; m < 4; ++m)
    store_tile(dst + m * 16 * ldd, ldd, acc[m], lane, bias, postmul, relu);
}

// ---------------- prologue kernels ----------------
__global__ void cvt_f32_to_bf16(const float* __restrict__ src,
                                __bf16* __restrict__ dst, long n) {
  long i = (long)blockIdx.x * blockDim.x + threadIdx.x;
  long stride = (long)gridDim.x * blockDim.x;
  for (; i < n; i += stride) dst[i] = (__bf16)src[i];
}

__global__ void scale_prep(const float* __restrict__ g,
                           float* __restrict__ scale,
                           float* __restrict__ slsum) {
  __shared__ float red[CDIM];
  int f = blockIdx.x, c = threadIdx.x;
  float gv = g[f * CDIM + c];
  float s  = 0.2f * log1pf(__expf(0.5f * gv));   // 0.2 * softplus(0.5 g)
  scale[f * CDIM + c] = s;
  red[c] = __logf(s);
  __syncthreads();
  for (int st = CDIM / 2; st > 0; st >>= 1) {
    if (c < st) red[c] += red[c + st];
    __syncthreads();
  }
  if (c == 0) slsum[f] = red[0];
}

// ---------------- fused flow kernel ----------------
__global__ __launch_bounds__(NTHR)
void affine_flow_kernel(const float*  __restrict__ z0,
                        const __bf16* __restrict__ Wp,
                        const float*  __restrict__ off,
                        const __bf16* __restrict__ w0,
                        const float*  __restrict__ b0,
                        const __bf16* __restrict__ wh,
                        const float*  __restrict__ bh,
                        const __bf16* __restrict__ wo,
                        const float*  __restrict__ bo,
                        const float*  __restrict__ scale,
                        const float*  __restrict__ slsum,
                        float*        __restrict__ out) {
  extern __shared__ __bf16 smem[];
  __bf16* xb = smem;                       // MT x 512  current activations
  __bf16* yb = xb + MT * CDIM;             // MT x 512  y / hidden-stage / 'a'
  __bf16* hb = yb + MT * CDIM;             // MT x 1024 subnet hidden (single buf)
  float*  ldb = (float*)(hb + MT * COUPW); // MT log-det partials
  // LDS total: 64KB + 64KB + 128KB + 256B = 256.25KB of 320KB WGP budget

  const int tid  = threadIdx.x;
  const int lane = tid & 31;
  const int cg   = tid >> 5;    // 0..7 : 64-column group; wave owns all 64 rows
  const int row0 = blockIdx.x * MT;

  // load z0 tile -> bf16 LDS
  for (int i = tid; i < MT * CDIM; i += NTHR)
    xb[i] = (__bf16)z0[row0 * CDIM + i];
  if (tid < MT) ldb[tid] = 0.f;
  __syncthreads();

  for (int f = 0; f < NFLOW; ++f) {
    // ---- y = x * scale + off ----
    {
      const float* sc = scale + f * CDIM;
      const float* of = off   + f * CDIM;
      for (int i = tid; i < MT * CDIM; i += NTHR) {
        int c = i & (CDIM - 1);
        yb[i] = (__bf16)((float)xb[i] * sc[c] + of[c]);
      }
    }
    __syncthreads();

    // ---- x = y @ Wp^T   (M=64, N=512, K=512) -> xb ----
    {
      const __bf16* W = Wp + (long)f * CDIM * CDIM;
      for (int t = 0; t < 4; ++t) {
        int nt = cg * 4 + t;
        v8f acc[4] = {{0.f},{0.f},{0.f},{0.f}};
        gemm_tile4(yb, CDIM, W + nt * 16 * CDIM, CDIM, CDIM, lane, acc);
        store_tile4(xb + nt * 16, CDIM, acc, lane, 0.f, 1.f, false);
      }
    }
    __syncthreads();

    // ---- h = relu(x1 @ w0^T + b0)  (N=1024, K=256) -> hb ----
    {
      const __bf16* W    = w0 + (long)f * COUPW * DD1;
      const float*  bias = b0 + f * COUPW;
      for (int t = 0; t < 8; ++t) {
        int nt = cg * 8 + t;
        v8f acc[4] = {{0.f},{0.f},{0.f},{0.f}};
        gemm_tile4(xb, CDIM, W + nt * 16 * DD1, DD1, DD1, lane, acc);
        store_tile4(hb + nt * 16, COUPW, acc, lane,
                    bias[nt * 16 + (lane & 15)], 1.f, true);
      }
    }
    __syncthreads();

    // ---- hidden layers (in-place, N=K=1024): h = relu(h @ wh[l]^T + bh[l])
    //      phase 0 (cols 0..511)  -> staged in yb
    //      phase 1 (cols 512..1023) -> held in registers across the barrier
    for (int l = 0; l < NBLK; ++l) {
      const __bf16* W    = wh + (long)(f * NBLK + l) * COUPW * COUPW;
      const float*  bias = bh + (f * NBLK + l) * COUPW;

      // phase 0: outputs go to yb (free), reads old hb
      for (int t = 0; t < 4; ++t) {
        int nt = cg * 4 + t;                    // cols 0..511
        v8f acc[4] = {{0.f},{0.f},{0.f},{0.f}};
        gemm_tile4(hb, COUPW, W + nt * 16 * COUPW, COUPW, COUPW, lane, acc);
        store_tile4(yb + nt * 16, CDIM, acc, lane,
                    bias[nt * 16 + (lane & 15)], 1.f, true);
      }
      // phase 1: hold results in registers (reads old hb)
      v8f hold[16];
#pragma unroll
      for (int i = 0; i < 16; ++i) hold[i] = (v8f){0.f};
      for (int t = 0; t < 4; ++t) {
        int nt = 32 + cg * 4 + t;               // cols 512..1023
        gemm_tile4(hb, COUPW, W + nt * 16 * COUPW, COUPW, COUPW, lane, &hold[t * 4]);
      }
      __syncthreads();   // everyone done reading old hb
      // commit phase 1 from registers
      for (int t = 0; t < 4; ++t) {
        int nt = 32 + cg * 4 + t;
        store_tile4(hb + nt * 16, COUPW, &hold[t * 4], lane,
                    bias[nt * 16 + (lane & 15)], 1.f, true);
      }
      // commit phase 0: copy yb -> hb[:, :512] (vectorized 16B moves)
      for (int i = tid; i < MT * CDIM / 8; i += NTHR) {
        int r = i >> 6;          // 64 v8bf per row
        int c = i & 63;
        *(v8bf*)(hb + r * COUPW + c * 8) = *(const v8bf*)(yb + r * CDIM + c * 8);
      }
      __syncthreads();
    }

    // ---- a = 0.1 * (h @ wo^T + bo)  (N=512, K=1024) -> yb ----
    {
      const __bf16* W    = wo + (long)f * (2 * DD2) * COUPW;
      const float*  bias = bo + f * (2 * DD2);
      for (int t = 0; t < 4; ++t) {
        int nt = cg * 4 + t;
        v8f acc[4] = {{0.f},{0.f},{0.f},{0.f}};
        gemm_tile4(hb, COUPW, W + nt * 16 * COUPW, COUPW, COUPW, lane, acc);
        store_tile4(yb + nt * 16, CDIM, acc, lane,
                    bias[nt * 16 + (lane & 15)], 0.1f, false);
      }
    }
    __syncthreads();

    // ---- coupling:  x2 = x2*exp(s) + t ;  ld += sum(s) ----
    for (int i = tid; i < MT * DD2; i += NTHR) {
      int m = i >> 8;            // DD2 == 256
      int j = i & (DD2 - 1);
      float s  = 2.0f * tanhf((float)yb[m * CDIM + j]);
      float t  = (float)yb[m * CDIM + DD2 + j];
      float x2 = (float)xb[m * CDIM + DD1 + j];
      xb[m * CDIM + DD1 + j] = (__bf16)(x2 * __expf(s) + t);
      atomicAdd(&ldb[m], s);
    }
    __syncthreads();
  }

  // ---- writeback: zk then log_det (+ Σ_f Σ_c log scale) ----
  float sls = 0.f;
#pragma unroll
  for (int f = 0; f < NFLOW; ++f) sls += slsum[f];
  for (int i = tid; i < MT * CDIM; i += NTHR)
    out[row0 * CDIM + i] = (float)xb[i];
  if (tid < MT)
    out[(long)BROWS * CDIM + row0 + tid] = ldb[tid] + sls;
}

// ---------------- launch ----------------
extern "C" void kernel_launch(void* const* d_in, const int* in_sizes, int n_in,
                              void* d_out, int out_size, void* d_ws, size_t ws_size,
                              hipStream_t stream) {
  const float* z0  = (const float*)d_in[0];
  const float* Wp  = (const float*)d_in[1];
  const float* g   = (const float*)d_in[2];
  const float* off = (const float*)d_in[3];
  const float* w0  = (const float*)d_in[4];
  const float* b0  = (const float*)d_in[5];
  const float* wh  = (const float*)d_in[6];
  const float* bh  = (const float*)d_in[7];
  const float* wo  = (const float*)d_in[8];
  const float* bo  = (const float*)d_in[9];
  float* out = (float*)d_out;

  const long nWp = (long)NFLOW * CDIM * CDIM;          // 2M
  const long nW0 = (long)NFLOW * COUPW * DD1;          // 2M
  const long nWh = (long)NFLOW * NBLK * COUPW * COUPW; // 16M
  const long nWo = (long)NFLOW * (2 * DD2) * COUPW;    // 4M

  char* ws = (char*)d_ws;
  __bf16* Wp_b = (__bf16*)ws;  ws += nWp * sizeof(__bf16);
  __bf16* w0_b = (__bf16*)ws;  ws += nW0 * sizeof(__bf16);
  __bf16* wh_b = (__bf16*)ws;  ws += nWh * sizeof(__bf16);
  __bf16* wo_b = (__bf16*)ws;  ws += nWo * sizeof(__bf16);
  float*  scl  = (float*)ws;   ws += (long)NFLOW * CDIM * sizeof(float);
  float*  sls  = (float*)ws;

  cvt_f32_to_bf16<<<2048, 256, 0, stream>>>(Wp, Wp_b, nWp);
  cvt_f32_to_bf16<<<2048, 256, 0, stream>>>(w0, w0_b, nW0);
  cvt_f32_to_bf16<<<4096, 256, 0, stream>>>(wh, wh_b, nWh);
  cvt_f32_to_bf16<<<2048, 256, 0, stream>>>(wo, wo_b, nWo);
  scale_prep<<<NFLOW, CDIM, 0, stream>>>(g, scl, sls);

  const size_t smem_bytes =
      (size_t)(2 * MT * CDIM + MT * COUPW) * sizeof(__bf16) + MT * sizeof(float);
  (void)hipFuncSetAttribute((const void*)affine_flow_kernel,
                            hipFuncAttributeMaxDynamicSharedMemorySize,
                            (int)smem_bytes);

  affine_flow_kernel<<<BROWS / MT, NTHR, smem_bytes, stream>>>(
      z0, Wp_b, off, w0_b, b0, wh_b, bh, wo_b, bo, scl, sls, out);
}
